// MinCutPoolLayerSparse_12300786336169
// MI455X (gfx1250) — matrix operand order repaired and checked by reading, not compiled, not run on hardware
//
#include <hip/hip_runtime.h>

#define NN 50000
#define DD 256
#define CC 512
#define EE 800000

typedef __attribute__((ext_vector_type(16))) __bf16 v16bf;
typedef __attribute__((ext_vector_type(8)))  __bf16 v8bf;
typedef __attribute__((ext_vector_type(8)))  float  v8f;

// build a 16-element bf16 fragment from two 16B-aligned 8-element chunks
static __device__ __forceinline__ v16bf ld8x2(const __bf16* p0, const __bf16* p1) {
  v8bf lo = *(const v8bf*)p0;
  v8bf hi = *(const v8bf*)p1;
  v16bf r;
#pragma unroll
  for (int i = 0; i < 8; ++i) { r[i] = lo[i]; r[i + 8] = hi[i]; }
  return r;
}

// ---------------------------------------------------------------- zero
__global__ void zero_kernel(float* __restrict__ p, long n) {
  long i = (long)blockIdx.x * 256 + threadIdx.x;
  if (i < n) p[i] = 0.f;
}

// ---------------------------------------------------------------- W[k][c] f32 -> Wt[c][k] bf16
__global__ void cvtT_kernel(const float* __restrict__ W, __bf16* __restrict__ Wt,
                            int K, int NC) {
  long i = (long)blockIdx.x * 256 + threadIdx.x;   // output index over NC*K
  if (i >= (long)K * NC) return;
  int c = (int)(i / K), k = (int)(i % K);
  Wt[i] = (__bf16)W[(size_t)k * NC + c];
}

// ---------------------------------------------------------------- row GEMM (+ optional fused softmax)
// Out[16 x NCOLS] = x[16 x 256] @ W[256 x NCOLS] + bias ; block = NCOLS threads
// Wt is pre-transposed bf16 [NCOLS][256]
template <int NCOLS, bool SOFTMAX>
__global__ __launch_bounds__(NCOLS)
void rowgemm_kernel(const float* __restrict__ X, const __bf16* __restrict__ Wt,
                    const float* __restrict__ bias, float* __restrict__ OutF,
                    __bf16* __restrict__ OutBf) {
  static_assert(!SOFTMAX || NCOLS == CC, "softmax path assumes 512 cols / 16 waves");
  __shared__ __attribute__((aligned(16))) __bf16 sx[16][DD + 16]; // 16B rows, few bank overlaps
  __shared__ float slog[SOFTMAX ? 16 : 1][CC];

  const int tid  = threadIdx.x;
  const int lane = tid & 31;
  const int wv   = tid >> 5;
  const int row0 = blockIdx.x * 16;

  for (int idx = tid; idx < 16 * DD; idx += NCOLS) {
    int r = idx >> 8, k = idx & (DD - 1);
    sx[r][k] = (__bf16)X[(size_t)(row0 + r) * DD + k];   // native cvt
  }
  __syncthreads();

  const int m    = lane & 15;
  const int kb   = (lane >> 4) * 8;    // unused; A frag uses kb2 like B
  const int kb2  = (lane >> 4) * 16;   // B-frag K group
  const int n    = lane & 15;
  const int col0 = wv * 32;

  const __bf16* wc0 = Wt + (size_t)(col0 + n) * DD;
  const __bf16* wc1 = Wt + (size_t)(col0 + 16 + n) * DD;

  v8f acc0 = {}; v8f acc1 = {};
#pragma unroll
  for (int k0 = 0; k0 < DD; k0 += 32) {
    v16bf a  = ld8x2(&sx[m][k0 + kb2], &sx[m][k0 + kb2 + 8]);
    v16bf b0 = ld8x2(wc0 + k0 + kb2, wc0 + k0 + kb2 + 8);
    v16bf b1 = ld8x2(wc1 + k0 + kb2, wc1 + k0 + kb2 + 8);
    acc0 = __builtin_amdgcn_wmma_f32_16x16x32_bf16(false, a, false, b0, (short)0, acc0, false, false);
    acc1 = __builtin_amdgcn_wmma_f32_16x16x32_bf16(false, a, false, b1, (short)0, acc1, false, false);
  }

  const float bn0 = bias[col0 + n];
  const float bn1 = bias[col0 + 16 + n];
  const int   mh  = (lane >> 4) * 8;   // C/D layout: VGPR r -> row r (+8 hi lanes)

  if (SOFTMAX) {
#pragma unroll
    for (int r = 0; r < 8; ++r) {
      slog[mh + r][col0 + n]      = acc0[r] + bn0;
      slog[mh + r][col0 + 16 + n] = acc1[r] + bn1;
    }
    __syncthreads();
    // wave wv owns row wv (16 waves == 16 rows), 16 cols per lane
    float vals[16];
    float mx = -3.4e38f;
#pragma unroll
    for (int j = 0; j < 16; ++j) { vals[j] = slog[wv][lane + 32 * j]; mx = fmaxf(mx, vals[j]); }
#pragma unroll
    for (int s = 16; s > 0; s >>= 1) mx = fmaxf(mx, __shfl_xor(mx, s, 32));
    float sum = 0.f;
#pragma unroll
    for (int j = 0; j < 16; ++j) { vals[j] = __expf(vals[j] - mx); sum += vals[j]; }
#pragma unroll
    for (int s = 16; s > 0; s >>= 1) sum += __shfl_xor(sum, s, 32);
    const float inv = 1.f / sum;
#pragma unroll
    for (int j = 0; j < 16; ++j) {
      float sv = vals[j] * inv;
      size_t o = (size_t)(row0 + wv) * CC + lane + 32 * j;
      OutF[o]  = sv;             // f32 S (kernel output)
      OutBf[o] = (__bf16)sv;     // bf16 S for the S^T GEMMs
    }
  } else {
#pragma unroll
    for (int r = 0; r < 8; ++r) {
      OutBf[(size_t)(row0 + mh + r) * NCOLS + col0 + n]      = (__bf16)(acc0[r] + bn0);
      OutBf[(size_t)(row0 + mh + r) * NCOLS + col0 + 16 + n] = (__bf16)(acc1[r] + bn1);
    }
  }
}

// ---------------------------------------------------------------- Sᵀ @ B  (split-K, f32 atomic reduce)
// Out[512 x nb] += Sbf[k0c:kend, :512]ᵀ @ B[k0c:kend, :nb]
// block = 128 threads (4 waves, 16x64 tile). A staged in LDS transposed; B read direct
// (each B element consumed by exactly one lane -> no reuse, no staging).
template <typename TB>
__global__ __launch_bounds__(128)
void gemmT_kernel(const __bf16* __restrict__ Sbf, const TB* __restrict__ B,
                  int nb, float* __restrict__ Out, int Ktot) {
  __shared__ __attribute__((aligned(16))) __bf16 sAt[16][32];   // [m][k] -> frag reads contiguous
  const int tid  = threadIdx.x;
  const int lane = tid & 31;
  const int wv   = tid >> 5;
  const int n0   = blockIdx.x * 64;
  const int m0   = blockIdx.y * 16;
  const int k0c  = blockIdx.z * 2048;
  const int kend = (k0c + 2048 < Ktot) ? (k0c + 2048) : Ktot;

  const int m = lane & 15, kb = (lane >> 4) * 8, kb2 = (lane >> 4) * 16, n = lane & 15;
  const int kkf = tid >> 2, mbf = (tid & 3) * 4;   // fill mapping: 4 contiguous m per thread
  const int nc = n0 + wv * 16 + n;

  v8f acc = {};
  for (int k0 = k0c; k0 < kend; k0 += 32) {
    const bool full = (k0 + 32 <= kend);           // uniform scalar branch
    const __bf16* sp = Sbf + (size_t)(k0 + kkf) * CC + m0 + mbf;
    if (full) {
      sAt[mbf + 0][kkf] = sp[0];
      sAt[mbf + 1][kkf] = sp[1];
      sAt[mbf + 2][kkf] = sp[2];
      sAt[mbf + 3][kkf] = sp[3];
    } else {
      const __bf16 z = (__bf16)0.f;
      const bool ok = (k0 + kkf) < kend;
      sAt[mbf + 0][kkf] = ok ? sp[0] : z;
      sAt[mbf + 1][kkf] = ok ? sp[1] : z;
      sAt[mbf + 2][kkf] = ok ? sp[2] : z;
      sAt[mbf + 3][kkf] = ok ? sp[3] : z;
    }
    __syncthreads();

    v16bf a = ld8x2(&sAt[m][kb2], &sAt[m][kb2 + 8]);
    v16bf b;
    if (full) {
#pragma unroll
      for (int i = 0; i < 16; ++i)
        b[i] = (__bf16)(float)B[(size_t)(k0 + kb2 + i) * nb + nc];
    } else {
#pragma unroll
      for (int i = 0; i < 16; ++i) {
        int k = k0 + kb2 + i;
        b[i] = (k < kend) ? (__bf16)(float)B[(size_t)k * nb + nc] : (__bf16)0.f;
      }
    }
    acc = __builtin_amdgcn_wmma_f32_16x16x32_bf16(false, a, false, b, (short)0, acc, false, false);
    __syncthreads();
  }
  const int mh = (lane >> 4) * 8;
#pragma unroll
  for (int r = 0; r < 8; ++r)
    atomicAdd(&Out[(size_t)(m0 + mh + r) * nb + n0 + wv * 16 + n], acc[r]);
}

// ---------------------------------------------------------------- SpMM: adjS[row] += val * S[col]
__global__ __launch_bounds__(256)
void spmm_kernel(const int* __restrict__ erow, const int* __restrict__ ecol,
                 const float* __restrict__ eval, const float* __restrict__ S,
                 float* __restrict__ adjS) {
  long t = (long)blockIdx.x * 256 + threadIdx.x;   // E * 128 work items, 4 cols each
  long e = t >> 7;
  if (e >= EE) return;
  int c4 = (int)(t & 127) * 4;
  float v = eval[e];
  int cN = ecol[e], rN = erow[e];
  const float2* sp = (const float2*)(S + (size_t)cN * CC + c4);  // 8B-aligned
  float2 s0 = sp[0], s1 = sp[1];
  float* dp = adjS + (size_t)rN * CC + c4;
  atomicAdd(dp + 0, v * s0.x);
  atomicAdd(dp + 1, v * s0.y);
  atomicAdd(dp + 2, v * s1.x);
  atomicAdd(dp + 3, v * s1.y);
}

// ---------------------------------------------------------------- vol = Σ edge_val (softmax rows sum to 1)
__global__ __launch_bounds__(256)
void vol_kernel(const float* __restrict__ eval, float* __restrict__ volp) {
  __shared__ float red[256];
  float acc = 0.f;
  for (long i = (long)blockIdx.x * 256 + threadIdx.x; i < EE; i += (long)gridDim.x * 256)
    acc += eval[i];
  red[threadIdx.x] = acc; __syncthreads();
  for (int s = 128; s > 0; s >>= 1) {
    if (threadIdx.x < s) red[threadIdx.x] += red[threadIdx.x + s];
    __syncthreads();
  }
  if (threadIdx.x == 0) atomicAdd(volp, red[0]);
}

// ---------------------------------------------------------------- losses: -trace/vol, ||SS-I||_F
__global__ __launch_bounds__(1024)
void losses_kernel(const float* __restrict__ adj_new, const float* __restrict__ SSm,
                   const float* __restrict__ volp, float* __restrict__ out_loss) {
  __shared__ float r1[1024], r2[1024];
  float cut = 0.f, fro = 0.f;
  for (int i = threadIdx.x; i < CC; i += 1024) cut += adj_new[(size_t)i * CC + i];
  for (int i = threadIdx.x; i < CC * CC; i += 1024) {
    float v = SSm[i] - ((i % (CC + 1)) == 0 ? 1.f : 0.f);
    fro += v * v;
  }
  r1[threadIdx.x] = cut; r2[threadIdx.x] = fro; __syncthreads();
  for (int s = 512; s > 0; s >>= 1) {
    if (threadIdx.x < s) { r1[threadIdx.x] += r1[threadIdx.x + s]; r2[threadIdx.x] += r2[threadIdx.x + s]; }
    __syncthreads();
  }
  if (threadIdx.x == 0) {
    out_loss[0] = -r1[0] / (volp[0] + 1e-9f);
    out_loss[1] = sqrtf(r2[0]);
  }
}

// ---------------------------------------------------------------- launch
extern "C" void kernel_launch(void* const* d_in, const int* in_sizes, int n_in,
                              void* d_out, int out_size, void* d_ws, size_t ws_size,
                              hipStream_t stream) {
  (void)in_sizes; (void)n_in; (void)out_size; (void)ws_size;
  const float* x    = (const float*)d_in[0];
  const int*   erow = (const int*)  d_in[1];
  const int*   ecol = (const int*)  d_in[2];
  const float* evl  = (const float*)d_in[3];
  const float* Wa   = (const float*)d_in[4];
  const float* ba   = (const float*)d_in[5];
  const float* Wp   = (const float*)d_in[6];
  const float* bp   = (const float*)d_in[7];

  float* out     = (float*)d_out;
  float* Z       = out;                 // [512,256]
  float* adj_new = out + 131072;        // [512,512]
  float* loss    = out + 393216;        // 2 scalars
  float* S       = out + 393218;        // [N,512] f32

  float*  ws    = (float*)d_ws;
  float*  adjS  = ws;                                    // [N,512] f32   102.4 MB
  __bf16* Sbf   = (__bf16*)(ws + 25600000);              // [N,512] bf16   51.2 MB
  __bf16* Xpbf  = (__bf16*)(ws + 25600000 + 12800000);   // [N,256] bf16   25.6 MB
  __bf16* WaT   = (__bf16*)(ws + 44800000);              // [512,256] bf16
  __bf16* WpT   = (__bf16*)(ws + 44865536);              // [256,256] bf16
  float*  SSm   = ws + 44898304;                         // [512,512] f32
  float*  volp  = SSm + (size_t)CC * CC;                 // 1

  // zero accumulation targets
  long nz = (long)NN * CC;
  zero_kernel<<<(unsigned)((nz + 255) / 256), 256, 0, stream>>>(adjS, nz);
  zero_kernel<<<(CC * CC + 1 + 255) / 256, 256, 0, stream>>>(SSm, CC * CC + 1);
  zero_kernel<<<(393218 + 255) / 256, 256, 0, stream>>>(out, 393218);

  // transpose+convert weights to bf16 (one-time, tiny)
  cvtT_kernel<<<(DD * CC + 255) / 256, 256, 0, stream>>>(Wa, WaT, DD, CC);
  cvtT_kernel<<<(DD * DD + 255) / 256, 256, 0, stream>>>(Wp, WpT, DD, DD);

  // S = softmax(x @ W_assign + b_assign)  -> S f32 + Sbf
  rowgemm_kernel<CC, true><<<NN / 16, CC, 0, stream>>>(x, WaT, ba, S, Sbf);
  // X_proj = x @ W_proj + b_proj          -> bf16 only (consumed only as WMMA B operand)
  rowgemm_kernel<DD, false><<<NN / 16, DD, 0, stream>>>(x, WpT, bp, nullptr, Xpbf);
  // vol = sum(edge_val)
  vol_kernel<<<1024, 256, 0, stream>>>(evl, volp);
  // adj_S = A @ S
  spmm_kernel<<<(unsigned)(((long)EE * 128) / 256), 256, 0, stream>>>(erow, ecol, evl, S, adjS);

  // Z = Sᵀ X_proj ; adj_new = Sᵀ adj_S ; SS = Sᵀ S   (split-K: 25 chunks of 2048)
  dim3 gZ(DD / 64, CC / 16, 25);
  gemmT_kernel<__bf16><<<gZ, 128, 0, stream>>>(Sbf, Xpbf, DD, Z, NN);
  dim3 gA(CC / 64, CC / 16, 25);
  gemmT_kernel<float><<<gA, 128, 0, stream>>>(Sbf, adjS, CC, adj_new, NN);
  gemmT_kernel<__bf16><<<gA, 128, 0, stream>>>(Sbf, Sbf, CC, SSm, NN);

  losses_kernel<<<1, 1024, 0, stream>>>(adj_new, SSm, volp, loss);
}
